// FusedMoE_20134806683922
// MI455X (gfx1250) — matrix operand order repaired and checked by reading, not compiled, not run on hardware
//
#include <hip/hip_runtime.h>
#include <hip/hip_bf16.h>
#include <math.h>

// Problem constants (match reference)
#define E_    16
#define K_    1024
#define N_    1024
#define BLK_  32
#define T_    1024
#define RT_   4            // row-tiles (of 16 tokens) per wave -> 64-token macro tile

typedef _Float16 v16h __attribute__((ext_vector_type(16)));
typedef _Float16 v8h  __attribute__((ext_vector_type(8)));
typedef float    v8f  __attribute__((ext_vector_type(8)));
typedef unsigned v8u  __attribute__((ext_vector_type(8)));

// ---------- branchless dequant helpers (integer-only, f16-exact) ----------

// One packed byte (two fp4 E2M1 codes) -> two f16 values packed in a u32.
// f16 high-byte table for magnitudes 0..7: 00,38,3C,3E,40,42,44,46
__device__ __forceinline__ unsigned fp4pair_to_h2(unsigned b) {
    unsigned mag0 = b & 7u;
    unsigned mag1 = (b >> 4) & 7u;
    unsigned sel0 = (mag0 & 4u) ? 0x46444240u : 0x3E3C3800u;
    unsigned sel1 = (mag1 & 4u) ? 0x46444240u : 0x3E3C3800u;
    unsigned h0 = ((sel0 >> ((mag0 & 3u) * 8u)) & 0xFFu) << 8;
    unsigned h1 = ((sel1 >> ((mag1 & 3u) * 8u)) & 0xFFu) << 8;
    h0 |= (b & 8u) << 12;      // sign of low nibble -> bit 15
    h1 |= (b & 128u) << 8;     // sign of high nibble -> bit 15
    return h0 | (h1 << 16);
}

// scale byte [7]=sign [6:3]=exp(bias7) [2:0]=mant -> f16 bits (exact).
// normal: 2^(e-7)*(1+m/8) -> f16 exp = e+8, mant = m<<7
// denorm (e==0): m * 2^-9 -> f16 high-byte table: 00,18,1C,1E,20,21,22,23
__device__ __forceinline__ unsigned scale_to_h(unsigned s) {
    unsigned e = (s >> 3) & 15u;
    unsigned m = s & 7u;
    unsigned nb   = ((e + 8u) << 10) | (m << 7);
    unsigned dsel = (m & 4u) ? 0x23222120u : 0x1E1C1800u;
    unsigned db   = ((dsel >> ((m & 3u) * 8u)) & 0xFFu) << 8;
    unsigned h = e ? nb : db;
    h |= (s >> 7) << 15;
    return h;
}

// Load+dequant one lane's slice of a 32x16 f16 B tile.
// wrow: packed fp4 row (one int per original uint8), srow: scale row (one int per uint8)
// Lane holds column n, k = k0 + half*16 + [0..15]  (contiguous, matches A)
__device__ __forceinline__ v16h load_deq_tile(const int* __restrict__ wrow,
                                              const int* __restrict__ srow,
                                              int k0, int half) {
    const int kk = k0 + half * 16;
    const int4 qa = *reinterpret_cast<const int4*>(wrow + (kk >> 1));
    const int4 qb = *reinterpret_cast<const int4*>(wrow + (kk >> 1) + 4);
    const unsigned sh = scale_to_h((unsigned)srow[kk >> 5]);
    const _Float16 sc = __builtin_bit_cast(_Float16, (unsigned short)sh);
    v8u p;
    p[0] = fp4pair_to_h2((unsigned)qa.x);
    p[1] = fp4pair_to_h2((unsigned)qa.y);
    p[2] = fp4pair_to_h2((unsigned)qa.z);
    p[3] = fp4pair_to_h2((unsigned)qa.w);
    p[4] = fp4pair_to_h2((unsigned)qb.x);
    p[5] = fp4pair_to_h2((unsigned)qb.y);
    p[6] = fp4pair_to_h2((unsigned)qb.z);
    p[7] = fp4pair_to_h2((unsigned)qb.w);
    v16h vals = __builtin_bit_cast(v16h, p);
    return vals * sc;          // 8x v_pk_mul_f16; exact product
}

// Lane's slice of 16x32 f16 A tile: row m = lane%16,
// k = k0 + half*16 + [0..15]   (contiguous, matches B)
__device__ __forceinline__ v16h load_a_tile(const _Float16* __restrict__ arow,
                                            int k0, int half) {
    const v8h lo = *reinterpret_cast<const v8h*>(arow + k0 + half * 16);
    const v8h hi = *reinterpret_cast<const v8h*>(arow + k0 + half * 16 + 8);
    v16h a;
#pragma unroll
    for (int i = 0; i < 8; ++i) { a[i] = lo[i]; a[8 + i] = hi[i]; }
    return a;
}

// ---------- kernels ----------
__global__ void moe_zero_counts(int* counts) {
    if (threadIdx.x < E_) counts[threadIdx.x] = 0;
}

__global__ void moe_cvt_x(const float* __restrict__ x, _Float16* __restrict__ x16, int n) {
    int i = blockIdx.x * blockDim.x + threadIdx.x;
    if (i < n) x16[i] = (_Float16)x[i];
}

__global__ void moe_router(const float* __restrict__ logits,
                           int* __restrict__ counts,
                           int* __restrict__ bucket,
                           float* __restrict__ pw) {
    int t = blockIdx.x * blockDim.x + threadIdx.x;
    if (t >= T_) return;
    float l[E_];
#pragma unroll
    for (int i = 0; i < E_; ++i) l[i] = logits[t * E_ + i];
    int i1 = 0; float v1 = l[0];
#pragma unroll
    for (int i = 1; i < E_; ++i) if (l[i] > v1) { v1 = l[i]; i1 = i; }
    int i2 = (i1 == 0) ? 1 : 0; float v2 = l[i2];
#pragma unroll
    for (int i = 0; i < E_; ++i) if (i != i1 && l[i] > v2) { v2 = l[i]; i2 = i; }
    // renormalized top-2 softmax weights (global denominator cancels)
    float e2 = __expf(v2 - v1);
    float w1 = 1.0f / (1.0f + e2);
    float w2 = e2 * w1;
    int s1 = atomicAdd(&counts[i1], 1);
    bucket[i1 * T_ + s1] = t * 2;
    pw[t * 2] = w1;
    int s2 = atomicAdd(&counts[i2], 1);
    bucket[i2 * T_ + s2] = t * 2 + 1;
    pw[t * 2 + 1] = w2;
}

// GEMM1 + bias + SiLU*up.  grid = (N/128, T/64, E), block = 256 (8 waves).
// Each wave: 64 tokens x 16 act columns; B tiles dequantized once, used by 4 WMMAs.
__global__ __launch_bounds__(256) void moe_gemm1(
    const _Float16* __restrict__ x16,
    const int* __restrict__ w13, const int* __restrict__ s13,
    const float* __restrict__ b13,
    const int* __restrict__ counts, const int* __restrict__ bucket,
    _Float16* __restrict__ act) {
    const int e    = blockIdx.z;
    const int cnt  = counts[e];
    const int row0 = blockIdx.y * (16 * RT_);
    if (row0 >= cnt) return;

    __shared__ int s_pair[16 * RT_];
    if (threadIdx.x < 16 * RT_) {
        int r = row0 + threadIdx.x;
        s_pair[threadIdx.x] = bucket[e * T_ + (r < cnt ? r : row0)];
    }
    __syncthreads();

    const int wave = threadIdx.x >> 5;
    const int lane = threadIdx.x & 31;
    const int half = lane >> 4;
    const int lm   = lane & 15;
    const int ncol = blockIdx.x * 128 + wave * 16 + lm;   // this lane's act column

    const _Float16* arow[RT_];
#pragma unroll
    for (int j = 0; j < RT_; ++j)
        arow[j] = x16 + (size_t)(s_pair[j * 16 + lm] >> 1) * K_;

    const int* gw = w13 + ((size_t)e * 2 * N_ + ncol) * (K_ / 2);
    const int* uw = gw + (size_t)N_ * (K_ / 2);
    const int* gs = s13 + ((size_t)e * 2 * N_ + ncol) * (K_ / BLK_);
    const int* us = gs + (size_t)N_ * (K_ / BLK_);

    v8f gacc[RT_] = {}; v8f uacc[RT_] = {};
#pragma unroll 1
    for (int k0 = 0; k0 < K_; k0 += 32) {
        v16h bg = load_deq_tile(gw, gs, k0, half);
        v16h bu = load_deq_tile(uw, us, k0, half);
#pragma unroll
        for (int j = 0; j < RT_; ++j) {
            v16h a = load_a_tile(arow[j], k0, half);
            gacc[j] = __builtin_amdgcn_wmma_f32_16x16x32_f16(false, a, false, bg, (short)0, gacc[j], false, false);
            uacc[j] = __builtin_amdgcn_wmma_f32_16x16x32_f16(false, a, false, bu, (short)0, uacc[j], false, false);
        }
    }
    const float gb = b13[e * 2 * N_ + ncol];
    const float ub = b13[e * 2 * N_ + N_ + ncol];
#pragma unroll
    for (int j = 0; j < RT_; ++j) {
#pragma unroll
        for (int r = 0; r < 8; ++r) {
            int m = j * 16 + r + 8 * half;    // C/D layout: VGPR r, lanes>=16 hold M=r+8
            if (row0 + m < cnt) {
                float g = gacc[j][r] + gb;
                float u = uacc[j][r] + ub;
                float s = g * __builtin_amdgcn_rcpf(1.0f + __expf(-g));
                act[(size_t)s_pair[m] * N_ + ncol] = (_Float16)(s * u);
            }
        }
    }
}

// GEMM2 + bias + per-pair router weight.  grid = (K/128, T/64, E), block = 256.
__global__ __launch_bounds__(256) void moe_gemm2(
    const _Float16* __restrict__ act,
    const int* __restrict__ w2, const int* __restrict__ s2,
    const float* __restrict__ b2,
    const int* __restrict__ counts, const int* __restrict__ bucket,
    const float* __restrict__ pw,
    float* __restrict__ obuf) {
    const int e    = blockIdx.z;
    const int cnt  = counts[e];
    const int row0 = blockIdx.y * (16 * RT_);
    if (row0 >= cnt) return;

    __shared__ int   s_pair[16 * RT_];
    __shared__ float s_w[16 * RT_];
    if (threadIdx.x < 16 * RT_) {
        int r = row0 + threadIdx.x;
        int p = bucket[e * T_ + (r < cnt ? r : row0)];
        s_pair[threadIdx.x] = p;
        s_w[threadIdx.x]    = pw[p];
    }
    __syncthreads();

    const int wave = threadIdx.x >> 5;
    const int lane = threadIdx.x & 31;
    const int half = lane >> 4;
    const int lm   = lane & 15;
    const int kcol = blockIdx.x * 128 + wave * 16 + lm;   // output K column

    const _Float16* arow[RT_];
#pragma unroll
    for (int j = 0; j < RT_; ++j)
        arow[j] = act + (size_t)s_pair[j * 16 + lm] * N_;

    const int* ww = w2 + ((size_t)e * K_ + kcol) * (N_ / 2);
    const int* ss = s2 + ((size_t)e * K_ + kcol) * (N_ / BLK_);

    v8f acc[RT_] = {};
#pragma unroll 2
    for (int n0 = 0; n0 < N_; n0 += 32) {
        v16h b = load_deq_tile(ww, ss, n0, half);
#pragma unroll
        for (int j = 0; j < RT_; ++j) {
            v16h a = load_a_tile(arow[j], n0, half);
            acc[j] = __builtin_amdgcn_wmma_f32_16x16x32_f16(false, a, false, b, (short)0, acc[j], false, false);
        }
    }
    const float bb = b2[e * K_ + kcol];
#pragma unroll
    for (int j = 0; j < RT_; ++j) {
#pragma unroll
        for (int r = 0; r < 8; ++r) {
            int m = j * 16 + r + 8 * half;
            if (row0 + m < cnt)
                obuf[(size_t)s_pair[m] * K_ + kcol] = (acc[j][r] + bb) * s_w[m];
        }
    }
}

__global__ void moe_reduce(const float* __restrict__ obuf, float* __restrict__ out, int n) {
    int i = blockIdx.x * blockDim.x + threadIdx.x;
    if (i < n) {
        int t = i >> 10;          // / K_
        int k = i & (K_ - 1);
        out[i] = obuf[((size_t)t * 2) * K_ + k] + obuf[((size_t)t * 2 + 1) * K_ + k];
    }
}

// ---------- launch ----------
extern "C" void kernel_launch(void* const* d_in, const int* in_sizes, int n_in,
                              void* d_out, int out_size, void* d_ws, size_t ws_size,
                              hipStream_t stream) {
    const float* x      = (const float*)d_in[0];   // (T,K) f32
    const float* logits = (const float*)d_in[1];   // (T,E) f32
    const int*   w13    = (const int*)d_in[2];     // (E,2N,K/2) packed fp4 bytes (as int32)
    const int*   s13    = (const int*)d_in[3];     // (E,2N,K/32)
    const int*   w2     = (const int*)d_in[4];     // (E,K,N/2)
    const int*   s2     = (const int*)d_in[5];     // (E,K,N/32)
    const float* b13    = (const float*)d_in[6];   // (E,2N)
    const float* b2     = (const float*)d_in[7];   // (E,K)
    float* out = (float*)d_out;

    // workspace layout
    char* ws = (char*)d_ws;
    int*      counts = (int*)ws;                                   // 256 B
    int*      bucket = (int*)(ws + 256);                           // E*T*4 = 64 KB
    float*    pw     = (float*)(ws + 256 + E_ * T_ * 4);           // 2T*4 = 8 KB
    _Float16* x16    = (_Float16*)(ws + 256 + E_ * T_ * 4 + 2 * T_ * 4);
    _Float16* act    = x16 + (size_t)T_ * K_;                      // 2T*N f16
    float*    obuf   = (float*)(act + (size_t)2 * T_ * N_);        // 2T*K f32

    hipLaunchKernelGGL(moe_zero_counts, dim3(1), dim3(32), 0, stream, counts);
    hipLaunchKernelGGL(moe_cvt_x, dim3((T_ * K_) / 256), dim3(256), 0, stream, x, x16, T_ * K_);
    hipLaunchKernelGGL(moe_router, dim3(T_ / 256), dim3(256), 0, stream, logits, counts, bucket, pw);
    hipLaunchKernelGGL(moe_gemm1, dim3(N_ / 128, T_ / (16 * RT_), E_), dim3(256), 0, stream,
                       x16, w13, s13, b13, counts, bucket, act);
    hipLaunchKernelGGL(moe_gemm2, dim3(K_ / 128, T_ / (16 * RT_), E_), dim3(256), 0, stream,
                       act, w2, s2, b2, counts, bucket, pw, obuf);
    hipLaunchKernelGGL(moe_reduce, dim3((T_ * K_) / 256), dim3(256), 0, stream, obuf, out, T_ * K_);
}
